// ResidualStack_53738630808177
// MI455X (gfx1250) — matrix operand order repaired and checked by reading, not compiled, not run on hardware
//
#include <hip/hip_runtime.h>
#include <hip/hip_bf16.h>

typedef __attribute__((ext_vector_type(16))) __bf16 v16bf;
typedef __attribute__((ext_vector_type(8)))  float  v8f;

#define N_BLOCKS 40
#define RC 128
#define DC 128
#define SC 256
#define CC 64
#define SKIP 4096
#define T_START 8188

// ---------------------------------------------------------------------------
// Prep kernels
// ---------------------------------------------------------------------------

__global__ void k_zero(float* __restrict__ p, int n) {
  int t = blockIdx.x * blockDim.x + threadIdx.x;
  if (t < n) p[t] = 0.f;
}

// Swizzle conv weights [N][DC][RC][2] -> bf16 A-fragments [N][tap][mt8][kt4][lane32][e16]
__global__ void k_swz_conv(const float* __restrict__ src, __bf16* __restrict__ dst, int total) {
  int tid = blockIdx.x * blockDim.x + threadIdx.x;
  if (tid >= total) return;
  int e    = tid & 15;
  int lane = (tid >> 4) & 31;
  int kt   = (tid >> 9) & 3;
  int mt   = (tid >> 11) & 7;
  int tap  = (tid >> 14) & 1;
  int i    = tid >> 15;
  int h    = lane >> 4;
  int row  = 16 * mt + (lane & 15);
  int col  = 32 * kt + 16 * h + e;
  float v = src[(((size_t)i * DC + row) * RC + col) * 2 + tap];
  dst[tid] = (__bf16)v;
}

// Swizzle projection weights [N][M][128] -> bf16 A-fragments [N][Mt][kt4][lane32][e16]
__global__ void k_swz_mat(const float* __restrict__ src, __bf16* __restrict__ dst,
                          int Mt, int total) {
  int tid = blockIdx.x * blockDim.x + threadIdx.x;
  if (tid >= total) return;
  int e    = tid & 15;
  int lane = (tid >> 4) & 31;
  int kt   = (tid >> 9) & 3;
  int rest = tid >> 11;
  int mt   = rest % Mt;
  int i    = rest / Mt;
  int h    = lane >> 4;
  int row  = 16 * mt + (lane & 15);
  int col  = 32 * kt + 16 * h + e;
  int M    = Mt * 16;
  dst[tid] = (__bf16)src[((size_t)i * M + row) * DC + col];
}

// cond_f[i][b][c] = cond[b] . cf_w[i][c] + cf_b[i][c]  (same for gate)
__global__ void k_cond(const float* __restrict__ cond,
                       const float* __restrict__ cfw, const float* __restrict__ cfb,
                       const float* __restrict__ cgw, const float* __restrict__ cgb,
                       float* __restrict__ condF, float* __restrict__ condG) {
  int tid = blockIdx.x * blockDim.x + threadIdx.x;
  if (tid >= N_BLOCKS * 2 * DC) return;
  int c = tid & 127;
  int b = (tid >> 7) & 1;
  int i = tid >> 8;
  const float* cb = cond + b * CC;
  const float* wf = cfw + ((size_t)i * DC + c) * CC;
  const float* wg = cgw + ((size_t)i * DC + c) * CC;
  float sf = 0.f, sg = 0.f;
  for (int k = 0; k < CC; ++k) { float cv = cb[k]; sf += cv * wf[k]; sg += cv * wg[k]; }
  condF[((size_t)i * 2 + b) * DC + c] = sf + cfb[(size_t)i * DC + c];
  condG[((size_t)i * 2 + b) * DC + c] = sg + cgb[(size_t)i * DC + c];
}

// ---------------------------------------------------------------------------
// Main per-block kernel: one wave (32 threads) per workgroup, one 16-step tile
// ---------------------------------------------------------------------------

__device__ __forceinline__ v16bf load_a16(const __bf16* p, int lane) {
  return *(const v16bf*)(p + (lane << 4));   // 32B contiguous per lane -> 2x b128
}

__device__ __forceinline__ v8f load_v8f(const float* p) {
  return *(const v8f*)p;                     // 32B-aligned contiguous -> 2x b128
}

__device__ __forceinline__ v8f wmma_bf16(v16bf a, v16bf b, v8f c) {
  return __builtin_amdgcn_wmma_f32_16x16x32_bf16(false, a, false, b, (short)0, c, false, false);
}

__global__ void __launch_bounds__(32) k_block(
    const float* __restrict__ xin,        // [2][128][T_in]
    float* __restrict__ xout,             // [2][128][L]
    const __bf16* __restrict__ fFr,       // [2][8][4][512]
    const __bf16* __restrict__ gFr,
    const __bf16* __restrict__ rFr,       // [8][4][512]
    const __bf16* __restrict__ sFr,       // [16][4][512]
    const float* __restrict__ condF,      // [2][128]
    const float* __restrict__ condG,
    const float* __restrict__ resB,       // [128]
    const float* __restrict__ skipB,      // [256]
    float* __restrict__ out,              // [2][256][4096], accumulated
    int T_in, int L, int d)
{
  __shared__ __bf16 abuf[DC * 16];        // activation tile (bf16) = 4KB, wave-private
  const int b    = blockIdx.y;
  const int lane = threadIdx.x & 31;
  const int t0   = blockIdx.x * 16;
  if (t0 >= L) return;                    // wave-uniform exit (EXEC stays full)
  const int h   = lane >> 4;
  const int col = lane & 15;
  const int t   = t0 + col;
  const bool valid = (t < L);
  const int tcl = valid ? t : (L - 1);    // clamp so gathers stay in-bounds
  const float* xb = xin + (size_t)b * RC * T_in;

  // ---- B operand fragments for the dilated conv: [tap][kt], K = channel ----
  v16bf bX[8];
  #pragma unroll
  for (int tap = 0; tap < 2; ++tap) {
    #pragma unroll
    for (int kt = 0; kt < 4; ++kt) {
      const float* p = xb + (size_t)(32 * kt + 16 * h) * T_in + (tcl + tap * d);
      v16bf r;
      #pragma unroll
      for (int e = 0; e < 16; ++e) r[e] = (__bf16)p[(size_t)e * T_in];
      bX[tap * 4 + kt] = r;
    }
  }

  const float* cF = condF + (size_t)b * DC;
  const float* cG = condG + (size_t)b * DC;

  // ---- filter + gate GEMMs, gating, stage activations to LDS (bf16) ----
  #pragma unroll
  for (int mt = 0; mt < 8; ++mt) {
    v8f f = load_v8f(cF + 16 * mt + 8 * h);   // 8 consecutive channels per lane
    v8f g = load_v8f(cG + 16 * mt + 8 * h);
    #pragma unroll
    for (int q = 0; q < 8; ++q) {             // q = tap*4 + kt
      int tap = q >> 2, kt = q & 3;
      size_t fo = ((size_t)(tap * 8 + mt) * 4 + kt) << 9;
      f = wmma_bf16(load_a16(fFr + fo, lane), bX[q], f);
      g = wmma_bf16(load_a16(gFr + fo, lane), bX[q], g);
    }
    #pragma unroll
    for (int r = 0; r < 8; ++r) {
      float tf = 2.f / (1.f + __expf(-2.f * f[r])) - 1.f;   // tanh, overflow-safe
      float sg = 1.f / (1.f + __expf(-g[r]));               // sigmoid
      int ch = 16 * mt + 8 * h + r;
      abuf[ch * 16 + col] = (__bf16)(tf * sg);
    }
  }
  // Same-wave LDS RAW: DScnt waits inserted by compiler; LDS region is wave-private.

  // ---- re-layout activations as B operand fragments (K = channel) ----
  v16bf bA[4];
  #pragma unroll
  for (int kt = 0; kt < 4; ++kt) {
    v16bf r;
    #pragma unroll
    for (int e = 0; e < 16; ++e)
      r[e] = abuf[(32 * kt + 16 * h + e) * 16 + col];
    bA[kt] = r;
  }

  // ---- residual projection + residual add -> xout ----
  #pragma unroll
  for (int mt = 0; mt < 8; ++mt) {
    v8f acc = load_v8f(resB + 16 * mt + 8 * h);
    #pragma unroll
    for (int r = 0; r < 8; ++r) {
      int ch = 16 * mt + 8 * h + r;
      acc[r] += xb[(size_t)ch * T_in + d + tcl];   // x tail offset = T_in - L = d
    }
    #pragma unroll
    for (int kt = 0; kt < 4; ++kt)
      acc = wmma_bf16(load_a16(rFr + (((size_t)mt * 4 + kt) << 9), lane), bA[kt], acc);
    if (valid) {
      #pragma unroll
      for (int r = 0; r < 8; ++r) {
        int ch = 16 * mt + 8 * h + r;
        xout[((size_t)b * RC + ch) * L + t] = acc[r];
      }
    }
  }

  // ---- skip projection -> accumulate into out (last SKIP timesteps) ----
  const int ts = t - (L - SKIP);
  const bool sval = valid && (ts >= 0);
  #pragma unroll
  for (int mt = 0; mt < 16; ++mt) {
    v8f acc = load_v8f(skipB + 16 * mt + 8 * h);
    #pragma unroll
    for (int kt = 0; kt < 4; ++kt)
      acc = wmma_bf16(load_a16(sFr + (((size_t)mt * 4 + kt) << 9), lane), bA[kt], acc);
    if (sval) {
      #pragma unroll
      for (int r = 0; r < 8; ++r) {
        int ch = 16 * mt + 8 * h + r;
        size_t oi = ((size_t)b * SC + ch) * SKIP + ts;
        out[oi] += acc[r];
      }
    }
  }
}

// ---------------------------------------------------------------------------
// Host launcher
// ---------------------------------------------------------------------------

extern "C" void kernel_launch(void* const* d_in, const int* in_sizes, int n_in,
                              void* d_out, int out_size, void* d_ws, size_t ws_size,
                              hipStream_t stream) {
  (void)in_sizes; (void)n_in; (void)out_size; (void)ws_size;
  const float* x      = (const float*)d_in[0];
  const float* cond   = (const float*)d_in[1];
  const float* filt_w = (const float*)d_in[2];
  const float* gate_w = (const float*)d_in[3];
  const float* cf_w   = (const float*)d_in[4];
  const float* cf_b   = (const float*)d_in[5];
  const float* cg_w   = (const float*)d_in[6];
  const float* cg_b   = (const float*)d_in[7];
  const float* res_w  = (const float*)d_in[8];
  const float* res_b  = (const float*)d_in[9];
  const float* skip_w = (const float*)d_in[10];
  const float* skip_b = (const float*)d_in[11];
  float* out = (float*)d_out;

  char* ws = (char*)d_ws;
  size_t off = 0;
  auto wsalloc = [&](size_t bytes) -> char* {
    off = (off + 255) & ~(size_t)255;
    char* p = ws + off;
    off += bytes;
    return p;
  };

  const size_t xBytes    = (size_t)2 * RC * 8192 * sizeof(float);
  const size_t convFrEls = (size_t)N_BLOCKS * 2 * 8 * 4 * 512;   // per conv array
  const size_t resFrEls  = (size_t)N_BLOCKS * 8 * 4 * 512;
  const size_t skipFrEls = (size_t)N_BLOCKS * 16 * 4 * 512;

  float*  xA    = (float*)wsalloc(xBytes);
  float*  xB    = (float*)wsalloc(xBytes);
  __bf16* fFr   = (__bf16*)wsalloc(convFrEls * 2);
  __bf16* gFr   = (__bf16*)wsalloc(convFrEls * 2);
  __bf16* rFr   = (__bf16*)wsalloc(resFrEls * 2);
  __bf16* sFr   = (__bf16*)wsalloc(skipFrEls * 2);
  float*  condF = (float*)wsalloc((size_t)N_BLOCKS * 2 * DC * sizeof(float));
  float*  condG = (float*)wsalloc((size_t)N_BLOCKS * 2 * DC * sizeof(float));

  // zero output accumulator (deterministic per call)
  {
    int n = 2 * SC * SKIP;
    k_zero<<<(n + 255) / 256, 256, 0, stream>>>(out, n);
  }
  // weight swizzles + conditioning
  {
    int n = (int)convFrEls;
    k_swz_conv<<<(n + 255) / 256, 256, 0, stream>>>(filt_w, fFr, n);
    k_swz_conv<<<(n + 255) / 256, 256, 0, stream>>>(gate_w, gFr, n);
  }
  {
    int n = (int)resFrEls;
    k_swz_mat<<<(n + 255) / 256, 256, 0, stream>>>(res_w, rFr, 8, n);
  }
  {
    int n = (int)skipFrEls;
    k_swz_mat<<<(n + 255) / 256, 256, 0, stream>>>(skip_w, sFr, 16, n);
  }
  {
    int n = N_BLOCKS * 2 * DC;
    k_cond<<<(n + 255) / 256, 256, 0, stream>>>(cond, cf_w, cf_b, cg_w, cg_b, condF, condG);
  }

  // sequential residual stack, ping-ponging x through workspace
  const float* cur = x;
  float* nxt = xA;
  int T = T_START;
  for (int i = 0; i < N_BLOCKS; ++i) {
    int dil = 1 << (i % 10);
    int Lo  = T - dil;
    int NT  = (Lo + 15) / 16;
    dim3 grid(NT, 2);
    k_block<<<grid, 32, 0, stream>>>(
        cur, nxt,
        fFr + (size_t)i * 2 * 8 * 4 * 512,
        gFr + (size_t)i * 2 * 8 * 4 * 512,
        rFr + (size_t)i * 8 * 4 * 512,
        sFr + (size_t)i * 16 * 4 * 512,
        condF + (size_t)i * 2 * DC,
        condG + (size_t)i * 2 * DC,
        res_b + (size_t)i * RC,
        skip_b + (size_t)i * SC,
        out, T, Lo, dil);
    cur = nxt;
    nxt = (nxt == xA) ? xB : xA;
    T = Lo;
  }
}
